// WsiEncoder_87479893884980
// MI455X (gfx1250) — matrix operand order repaired and checked by reading, not compiled, not run on hardware
//
#include <hip/hip_runtime.h>

#define P_NODES 30000
#define PPAD    30080      // multiple of 128 (block row coverage)
#define DIN     1536
#define DOUT    256
#define NEDGE   320000
#define ETOT    (NEDGE + P_NODES)
#define NEG_SLOPE 0.2f

typedef __attribute__((ext_vector_type(16))) __bf16 v16bf;
typedef __attribute__((ext_vector_type(8)))  float  v8f;

// ---------------------------------------------------------------- WMMA GEMM
// C[M x N] (f32, ldc, +col0) = A[M x K] (bf16 row-major) * Bt[N x K]^T (bf16) + bias[N]
// wave tile: 32 (M) x 64 (N); block = 4 waves -> 128 x 64
__global__ __launch_bounds__(128) void gemm_bf16_wmma(
    const __bf16* __restrict__ A, const __bf16* __restrict__ Bt,
    const float* __restrict__ bias, float* __restrict__ C,
    int M, int K, int ldc, int col0)
{
  const int lane   = threadIdx.x & 31;
  const int wave   = threadIdx.x >> 5;
  const int m_base = (blockIdx.x * 4 + wave) * 32;
  const int n_base = blockIdx.y * 64;
  if (m_base >= M) return;            // uniform per wave -> EXEC stays all-1s

  v8f acc[2][4] = {};

  const int arow = lane & 15;
  const int aoff = (lane >> 4) * 8;   // A: upper half-wave starts at K+8
  const int boff = (lane >> 4) * 8;   // B: upper half-wave starts at K+8
  const int bcol = lane & 15;

  for (int k0 = 0; k0 < K; k0 += 32) {
    v16bf a[2], b[4];
#pragma unroll
    for (int i = 0; i < 2; i++) {
      const __bf16* p = A + (size_t)(m_base + i * 16 + arow) * K + k0 + aoff;
      union { uint4 q[2]; v16bf v; } u;
      u.q[0] = *(const uint4*)p;            // K = k0+off .. +7
      u.q[1] = *(const uint4*)(p + 16);     // K = k0+off+16 .. +7
      a[i] = u.v;
      __builtin_prefetch(p + 32, 0, 3);     // global_prefetch next K-step
    }
#pragma unroll
    for (int j = 0; j < 4; j++) {
      const __bf16* p = Bt + (size_t)(n_base + j * 16 + bcol) * K + k0 + boff;
      union { uint4 q[2]; v16bf v; } u;
      u.q[0] = *(const uint4*)p;            // K = k0+off .. +7
      u.q[1] = *(const uint4*)(p + 16);     // K = k0+off+16 .. +7
      b[j] = u.v;
    }
#pragma unroll
    for (int i = 0; i < 2; i++)
#pragma unroll
      for (int j = 0; j < 4; j++)
        acc[i][j] = __builtin_amdgcn_wmma_f32_16x16x32_bf16(
            false, a[i], false, b[j], (short)0, acc[i][j], false, false);
  }

  const int crow = (lane >> 4) * 8;
#pragma unroll
  for (int i = 0; i < 2; i++)
#pragma unroll
    for (int j = 0; j < 4; j++) {
      const int col_local = n_base + j * 16 + (lane & 15);
      const float bv = bias[col_local];
      float* cp = C + (size_t)(m_base + i * 16 + crow) * ldc + col0 + col_local;
#pragma unroll
      for (int r = 0; r < 8; r++) cp[(size_t)r * ldc] = acc[i][j][r] + bv;
    }
}

// ---------------------------------------------------------------- prep kernels
__global__ void cvt_pad_bf16_k(const float* __restrict__ src, __bf16* __restrict__ dst,
                               long long realElems, long long totElems)
{
  long long i = (long long)blockIdx.x * blockDim.x + threadIdx.x;
  if (i >= totElems) return;
  dst[i] = (i < realElems) ? (__bf16)src[i] : (__bf16)0.0f;
}

__global__ void transpose_w_bf16_k(const float* __restrict__ src, __bf16* __restrict__ dst,
                                   int K, int N, int row_off, int ldk)
{
  int i = blockIdx.x * blockDim.x + threadIdx.x;
  if (i >= K * N) return;
  int k = i / N, n = i % N;
  dst[(size_t)(row_off + n) * ldk + k] = (__bf16)src[i];
}

__global__ void copy_f32_k(const float* __restrict__ s, float* __restrict__ d, int n)
{
  int i = blockIdx.x * blockDim.x + threadIdx.x;
  if (i < n) d[i] = s[i];
}

// ---------------------------------------------------------------- edge phase
__device__ inline unsigned ordFromFloat(float f) {
  unsigned u = __float_as_uint(f);
  return (u & 0x80000000u) ? ~u : (u | 0x80000000u);
}
__device__ inline float floatFromOrd(unsigned u) {
  return __uint_as_float((u & 0x80000000u) ? (u & 0x7fffffffu) : ~u);
}
__device__ inline float lterm(float m, float a) {
  m = m > 0.f ? m : NEG_SLOPE * m;
  return m * a;
}

// one wave per edge; lanes 0-15 = head 0 (ch 0-127), lanes 16-31 = head 1
__global__ __launch_bounds__(256) void edge_scores_k(
    const float* __restrict__ XL, const float* __restrict__ XR, int ld,
    const int* __restrict__ srcA, const int* __restrict__ dstA,
    const float* __restrict__ att, float* __restrict__ ebuf,
    unsigned* __restrict__ emax)
{
  int e = blockIdx.x * 8 + (threadIdx.x >> 5);
  if (e >= ETOT) return;
  int lane = threadIdx.x & 31;
  int s, d;
  if (e < NEDGE) { s = srcA[e]; d = dstA[e]; } else { s = e - NEDGE; d = s; }
  int ch = lane * 8;
  const float4* pl = (const float4*)(XL + (size_t)s * ld + ch);
  const float4* pr = (const float4*)(XR + (size_t)d * ld + ch);
  const float4* pa = (const float4*)(att + ch);
  float acc = 0.f;
#pragma unroll
  for (int t = 0; t < 2; t++) {
    float4 l = pl[t], r = pr[t], a = pa[t];
    acc += lterm(l.x + r.x, a.x) + lterm(l.y + r.y, a.y) +
           lterm(l.z + r.z, a.z) + lterm(l.w + r.w, a.w);
  }
#pragma unroll
  for (int k = 1; k < 16; k <<= 1) acc += __shfl_xor(acc, k, 32);  // per-head reduce
  if ((lane & 15) == 0) {
    int h = lane >> 4;
    ebuf[(size_t)e * 2 + h] = acc;
    atomicMax(&emax[d * 2 + h], ordFromFloat(acc));
  }
}

__global__ __launch_bounds__(256) void edge_expsum_k(
    const float* __restrict__ ebuf, const int* __restrict__ dstA,
    const unsigned* __restrict__ emax, float* __restrict__ exbuf,
    float* __restrict__ den)
{
  int i = blockIdx.x * blockDim.x + threadIdx.x;
  if (i >= ETOT * 2) return;
  int e = i >> 1, h = i & 1;
  int d = (e < NEDGE) ? dstA[e] : (e - NEDGE);
  float mx = floatFromOrd(emax[d * 2 + h]);
  float ex = __expf(ebuf[i] - mx);
  exbuf[i] = ex;
  atomicAdd(&den[d * 2 + h], ex);
}

__global__ __launch_bounds__(256) void edge_aggregate_k(
    const float* __restrict__ XL, int ld,
    const float* __restrict__ exbuf, const float* __restrict__ den,
    const int* __restrict__ srcA, const int* __restrict__ dstA,
    float* __restrict__ agg)
{
  int e = blockIdx.x * 8 + (threadIdx.x >> 5);
  if (e >= ETOT) return;
  int lane = threadIdx.x & 31;
  int s, d;
  if (e < NEDGE) { s = srcA[e]; d = dstA[e]; } else { s = e - NEDGE; d = s; }
  int h = lane >> 4;
  float alpha = exbuf[(size_t)e * 2 + h] / den[d * 2 + h];
  int ch = lane * 8;
  const float* pl = XL + (size_t)s * ld + ch;
  float* pd = agg + (size_t)d * DOUT + ch;
#pragma unroll
  for (int t = 0; t < 8; t++) atomicAdd(&pd[t], alpha * pl[t]);
}

// ---------------------------------------------------------------- node phase
// one wave per node: LN over 256 channels (8 per lane), ELU, +res, emit f32+bf16
__global__ __launch_bounds__(256) void node_post1_k(
    const float* __restrict__ agg, const float* __restrict__ bias,
    const float* __restrict__ g, const float* __restrict__ b,
    const float* __restrict__ res, int res_ld,
    float* __restrict__ h1, __bf16* __restrict__ h1bf)
{
  int node = blockIdx.x * 8 + (threadIdx.x >> 5);
  if (node >= PPAD) return;
  int lane = threadIdx.x & 31;
  int ch = lane * 8;
  if (node >= P_NODES) {                   // zero pad rows feeding GEMM2
    __bf16* pb = h1bf + (size_t)node * DOUT + ch;
#pragma unroll
    for (int t = 0; t < 8; t++) pb[t] = (__bf16)0.0f;
    return;
  }
  float v[8];
  const float* pa = agg + (size_t)node * DOUT + ch;
#pragma unroll
  for (int t = 0; t < 8; t++) v[t] = pa[t] + bias[ch + t];
  float s = 0.f;
#pragma unroll
  for (int t = 0; t < 8; t++) s += v[t];
#pragma unroll
  for (int k = 1; k < 32; k <<= 1) s += __shfl_xor(s, k, 32);
  float mu = s * (1.0f / DOUT);
  float q = 0.f;
#pragma unroll
  for (int t = 0; t < 8; t++) { float dv = v[t] - mu; q += dv * dv; }
#pragma unroll
  for (int k = 1; k < 32; k <<= 1) q += __shfl_xor(q, k, 32);
  float rinv = rsqrtf(q * (1.0f / DOUT) + 1e-5f);
  const float* pres = res + (size_t)node * res_ld + ch;
  float* ph = h1 + (size_t)node * DOUT + ch;
  __bf16* pb = h1bf + (size_t)node * DOUT + ch;
#pragma unroll
  for (int t = 0; t < 8; t++) {
    float ln = (v[t] - mu) * rinv * g[ch + t] + b[ch + t];
    float el = ln > 0.f ? ln : (__expf(ln) - 1.0f);
    float hv = el + pres[t];
    ph[t] = hv;
    pb[t] = (__bf16)hv;
  }
}

__global__ __launch_bounds__(256) void node_post2_k(
    const float* __restrict__ agg, const float* __restrict__ bias,
    const float* __restrict__ g, const float* __restrict__ b,
    const float* __restrict__ h1, float* __restrict__ out)
{
  int node = blockIdx.x * 8 + (threadIdx.x >> 5);
  if (node >= P_NODES) return;
  int lane = threadIdx.x & 31;
  int ch = lane * 8;
  float v[8];
  const float* pa = agg + (size_t)node * DOUT + ch;
#pragma unroll
  for (int t = 0; t < 8; t++) v[t] = pa[t] + bias[ch + t];
  float s = 0.f;
#pragma unroll
  for (int t = 0; t < 8; t++) s += v[t];
#pragma unroll
  for (int k = 1; k < 32; k <<= 1) s += __shfl_xor(s, k, 32);
  float mu = s * (1.0f / DOUT);
  float q = 0.f;
#pragma unroll
  for (int t = 0; t < 8; t++) { float dv = v[t] - mu; q += dv * dv; }
#pragma unroll
  for (int k = 1; k < 32; k <<= 1) q += __shfl_xor(q, k, 32);
  float rinv = rsqrtf(q * (1.0f / DOUT) + 1e-5f);
  const float* ph1 = h1 + (size_t)node * DOUT + ch;
  float* po = out + (size_t)node * DOUT + ch;
#pragma unroll
  for (int t = 0; t < 8; t++) {
    float ln = (v[t] - mu) * rinv * g[ch + t] + b[ch + t];
    float el = ln > 0.f ? ln : (__expf(ln) - 1.0f);
    float h1v = ph1[t];
    po[t] = fmaxf(h1v, el + h1v);           // JK-max of h1 and h2
  }
}

// ---------------------------------------------------------------- launcher
extern "C" void kernel_launch(void* const* d_in, const int* in_sizes, int n_in,
                              void* d_out, int out_size, void* d_ws, size_t ws_size,
                              hipStream_t stream)
{
  const float* x     = (const float*)d_in[0];
  const int*   eidx  = (const int*)d_in[1];
  const int*   srcA  = eidx;
  const int*   dstA  = eidx + NEDGE;
  const float* W1l   = (const float*)d_in[2];
  const float* b1l   = (const float*)d_in[3];
  const float* W1r   = (const float*)d_in[4];
  const float* b1r   = (const float*)d_in[5];
  const float* att1  = (const float*)d_in[6];
  const float* bias1 = (const float*)d_in[7];
  const float* W2l   = (const float*)d_in[8];
  const float* b2l   = (const float*)d_in[9];
  const float* W2r   = (const float*)d_in[10];
  const float* b2r   = (const float*)d_in[11];
  const float* att2  = (const float*)d_in[12];
  const float* bias2 = (const float*)d_in[13];
  const float* ln1_g = (const float*)d_in[14];
  const float* ln1_b = (const float*)d_in[15];
  const float* ln2_g = (const float*)d_in[16];
  const float* ln2_b = (const float*)d_in[17];
  const float* resW  = (const float*)d_in[18];
  const float* resb  = (const float*)d_in[19];
  (void)in_sizes; (void)n_in; (void)out_size; (void)ws_size;

  char* ws = (char*)d_ws;
  size_t off = 0;
  auto alloc = [&](size_t bytes) -> char* {
    char* p = ws + off;
    off += (bytes + 255) & ~(size_t)255;
    return p;
  };
  __bf16*   x_bf   = (__bf16*)  alloc((size_t)PPAD * DIN * 2);
  __bf16*   Bt1    = (__bf16*)  alloc((size_t)768 * DIN * 2);
  float*    biasc1 = (float*)   alloc(768 * 4);
  float*    out1   = (float*)   alloc((size_t)PPAD * 768 * 4);   // [XL1|XR1|RES]
  float*    ebuf   = (float*)   alloc((size_t)ETOT * 2 * 4);
  float*    exbuf  = (float*)   alloc((size_t)ETOT * 2 * 4);
  unsigned* emax   = (unsigned*)alloc((size_t)P_NODES * 2 * 4);
  float*    den    = (float*)   alloc((size_t)P_NODES * 2 * 4);
  float*    agg    = (float*)   alloc((size_t)P_NODES * DOUT * 4);
  float*    h1     = (float*)   alloc((size_t)PPAD * DOUT * 4);
  __bf16*   h1bf   = (__bf16*)  alloc((size_t)PPAD * DOUT * 2);
  __bf16*   Bt2    = (__bf16*)  alloc((size_t)512 * DOUT * 2);
  float*    biasc2 = (float*)   alloc(512 * 4);
  float*    out2   = (float*)   alloc((size_t)PPAD * 512 * 4);   // [XL2|XR2]

  // -- prep: x -> bf16 (padded), weights -> transposed bf16, bias concat
  {
    long long tot = (long long)PPAD * DIN, real = (long long)P_NODES * DIN;
    cvt_pad_bf16_k<<<(unsigned)((tot + 255) / 256), 256, 0, stream>>>(x, x_bf, real, tot);
  }
  {
    int n = DIN * DOUT, blk = (n + 255) / 256;
    transpose_w_bf16_k<<<blk, 256, 0, stream>>>(W1l,  Bt1, DIN, DOUT, 0,   DIN);
    transpose_w_bf16_k<<<blk, 256, 0, stream>>>(W1r,  Bt1, DIN, DOUT, 256, DIN);
    transpose_w_bf16_k<<<blk, 256, 0, stream>>>(resW, Bt1, DIN, DOUT, 512, DIN);
  }
  copy_f32_k<<<1, 256, 0, stream>>>(b1l,  biasc1,       256);
  copy_f32_k<<<1, 256, 0, stream>>>(b1r,  biasc1 + 256, 256);
  copy_f32_k<<<1, 256, 0, stream>>>(resb, biasc1 + 512, 256);

  // -- fused GEMM1: out1 = x @ [W1l|W1r|resW] + [b1l|b1r|resb]
  {
    dim3 grid(PPAD / 128, 768 / 64);
    gemm_bf16_wmma<<<grid, 128, 0, stream>>>(x_bf, Bt1, biasc1, out1, PPAD, DIN, 768, 0);
  }

  // -- layer 1 attention (segment softmax over incoming edges) + aggregate
  hipMemsetAsync(emax, 0, (size_t)P_NODES * 2 * 4, stream);
  hipMemsetAsync(den,  0, (size_t)P_NODES * 2 * 4, stream);
  hipMemsetAsync(agg,  0, (size_t)P_NODES * DOUT * 4, stream);
  edge_scores_k<<<(ETOT + 7) / 8, 256, 0, stream>>>(out1, out1 + 256, 768, srcA, dstA, att1, ebuf, emax);
  edge_expsum_k<<<(ETOT * 2 + 255) / 256, 256, 0, stream>>>(ebuf, dstA, emax, exbuf, den);
  edge_aggregate_k<<<(ETOT + 7) / 8, 256, 0, stream>>>(out1, 768, exbuf, den, srcA, dstA, agg);
  node_post1_k<<<(PPAD + 7) / 8, 256, 0, stream>>>(agg, bias1, ln1_g, ln1_b, out1 + 512, 768, h1, h1bf);

  // -- layer 2 prep + GEMM: out2 = h1 @ [W2l|W2r] + [b2l|b2r]
  {
    int n = DOUT * DOUT, blk = (n + 255) / 256;
    transpose_w_bf16_k<<<blk, 256, 0, stream>>>(W2l, Bt2, DOUT, DOUT, 0,   DOUT);
    transpose_w_bf16_k<<<blk, 256, 0, stream>>>(W2r, Bt2, DOUT, DOUT, 256, DOUT);
  }
  copy_f32_k<<<1, 256, 0, stream>>>(b2l, biasc2,       256);
  copy_f32_k<<<1, 256, 0, stream>>>(b2r, biasc2 + 256, 256);
  {
    dim3 grid(PPAD / 128, 512 / 64);
    gemm_bf16_wmma<<<grid, 128, 0, stream>>>(h1bf, Bt2, biasc2, out2, PPAD, DOUT, 512, 0);
  }

  // -- layer 2 attention + aggregate + final (LN, ELU, +h1, JK-max)
  hipMemsetAsync(emax, 0, (size_t)P_NODES * 2 * 4, stream);
  hipMemsetAsync(den,  0, (size_t)P_NODES * 2 * 4, stream);
  hipMemsetAsync(agg,  0, (size_t)P_NODES * DOUT * 4, stream);
  edge_scores_k<<<(ETOT + 7) / 8, 256, 0, stream>>>(out2, out2 + 256, 512, srcA, dstA, att2, ebuf, emax);
  edge_expsum_k<<<(ETOT * 2 + 255) / 256, 256, 0, stream>>>(ebuf, dstA, emax, exbuf, den);
  edge_aggregate_k<<<(ETOT + 7) / 8, 256, 0, stream>>>(out2, 512, exbuf, den, srcA, dstA, agg);
  node_post2_k<<<(P_NODES + 7) / 8, 256, 0, stream>>>(agg, bias2, ln2_g, ln2_b, h1, (float*)d_out);
}
